// QuantizedMatMul_22393959481942
// MI455X (gfx1250) — 8-bit matrix operand order repaired and checked by reading, compiled, not run on hardware
//
#include <hip/hip_runtime.h>

typedef __attribute__((ext_vector_type(8)))  int   v8i;
typedef __attribute__((ext_vector_type(4)))  int   v4i;
typedef __attribute__((ext_vector_type(2)))  int   v2i;
typedef __attribute__((ext_vector_type(8)))  float v8f;

#define BATCHES 64
#define MDIM    2048
#define KDIM    64
#define NDIM    2048
#define E4M3_MAX 448.0f

__global__ void init_amax(float* amax) {
    if (threadIdx.x < 2) amax[threadIdx.x] = 0.0f;
}

__global__ void amax_kernel(const float* __restrict__ in, float* __restrict__ out, int n4) {
    float m = 0.0f;
    for (int i = blockIdx.x * blockDim.x + threadIdx.x; i < n4; i += gridDim.x * blockDim.x) {
        float4 x = ((const float4*)in)[i];
        m = fmaxf(m, fmaxf(fmaxf(fabsf(x.x), fabsf(x.y)), fmaxf(fabsf(x.z), fabsf(x.w))));
    }
    for (int off = 16; off > 0; off >>= 1)
        m = fmaxf(m, __shfl_xor(m, off, 32));
    __shared__ float sm[8];
    if ((threadIdx.x & 31) == 0) sm[threadIdx.x >> 5] = m;
    __syncthreads();
    if (threadIdx.x == 0) {
        float bm = sm[0];
        for (int w = 1; w < (int)(blockDim.x >> 5); ++w) bm = fmaxf(bm, sm[w]);
        atomicMax((unsigned int*)out, __float_as_uint(bm));
    }
}

__global__ void quant_a(const float* __restrict__ in, unsigned int* __restrict__ aq,
                        const float* __restrict__ amax, int n4) {
    float am = amax[0];
    float rs = am > 0.0f ? (E4M3_MAX / am) : 1.0f;
    for (int i = blockIdx.x * blockDim.x + threadIdx.x; i < n4; i += gridDim.x * blockDim.x) {
        float4 x = ((const float4*)in)[i];
        int lo = __builtin_amdgcn_cvt_pk_fp8_f32(x.x * rs, x.y * rs, 0,  false);
        int pk = __builtin_amdgcn_cvt_pk_fp8_f32(x.z * rs, x.w * rs, lo, true);
        aq[i] = (unsigned int)pk;
    }
}

#define BT_ROWSTRIDE 80

__global__ void quant_b_transpose(const float* __restrict__ in, unsigned char* __restrict__ bq,
                                  const float* __restrict__ amax) {
    __shared__ unsigned char smem[64 * BT_ROWSTRIDE];
    const int b  = blockIdx.y;
    const int n0 = blockIdx.x * 64;
    float am = amax[0];
    float rs = am > 0.0f ? (E4M3_MAX / am) : 1.0f;

    for (int rep = 0; rep < 4; ++rep) {
        int unit = threadIdx.x + rep * 256;
        int k  = unit >> 4;
        int n4 = unit & 15;
        const float* src = in + ((size_t)(b * KDIM + k)) * NDIM + n0 + n4 * 4;
        float4 x = *(const float4*)src;
        int lo = __builtin_amdgcn_cvt_pk_fp8_f32(x.x * rs, x.y * rs, 0,  false);
        int pk = __builtin_amdgcn_cvt_pk_fp8_f32(x.z * rs, x.w * rs, lo, true);
        unsigned int u = (unsigned int)pk;
        #pragma unroll
        for (int j = 0; j < 4; ++j)
            smem[(n4 * 4 + j) * BT_ROWSTRIDE + k] = (unsigned char)((u >> (8 * j)) & 0xFF);
    }
    __syncthreads();

    int n  = threadIdx.x >> 2;
    int ch = threadIdx.x & 3;
    v4i v = *(const v4i*)(smem + n * BT_ROWSTRIDE + ch * 16);
    *(v4i*)(bq + ((size_t)(b * NDIM + n0 + n)) * KDIM + ch * 16) = v;
}

__global__ void __launch_bounds__(256)
gemm_fp8_wmma(const unsigned char* __restrict__ aq, const unsigned char* __restrict__ bq,
              float* __restrict__ out, const float* __restrict__ amax) {
    const int lane = threadIdx.x & 31;
    const int wave = threadIdx.x >> 5;
    const int b    = blockIdx.z;
    const int m0   = blockIdx.x * 128 + (wave & 3) * 32;
    const int n0   = blockIdx.y * 128 + (wave >> 2) * 64;

    const int hi = lane >> 4;
    const int lr = lane & 15;

    float a1 = amax[0], a2 = amax[1];
    float s1 = a1 > 0.0f ? a1 * (1.0f / E4M3_MAX) : 1.0f;
    float s2 = a2 > 0.0f ? a2 * (1.0f / E4M3_MAX) : 1.0f;
    const float dq = s1 * s2;

    const size_t abase = (size_t)b * MDIM * KDIM;
    const size_t bbase = (size_t)b * NDIM * KDIM;
    const size_t obase = (size_t)b * MDIM * NDIM;

    v8i afrag[2];
    #pragma unroll
    for (int t = 0; t < 2; ++t) {
        const unsigned char* ar = aq + abase + (size_t)(m0 + t * 16 + lr) * KDIM + hi * 16;
        v2i p0 = *(const v2i*)(ar +  0);
        v2i p1 = *(const v2i*)(ar +  8);
        v2i p2 = *(const v2i*)(ar + 32);
        v2i p3 = *(const v2i*)(ar + 40);
        v8i a;
        a[0] = p0.x; a[1] = p0.y; a[2] = p1.x; a[3] = p1.y;
        a[4] = p2.x; a[5] = p2.y; a[6] = p3.x; a[7] = p3.y;
        afrag[t] = a;
    }

    v8i bfrag[4];
    #pragma unroll
    for (int u = 0; u < 4; ++u) {
        const unsigned char* br = bq + bbase + (size_t)(n0 + u * 16 + lr) * KDIM + hi * 16;
        v4i q0 = *(const v4i*)(br);
        v4i q1 = *(const v4i*)(br + 32);
        v8i bb;
        bb[0] = q0.x; bb[1] = q0.y; bb[2] = q0.z; bb[3] = q0.w;
        bb[4] = q1.x; bb[5] = q1.y; bb[6] = q1.z; bb[7] = q1.w;
        bfrag[u] = bb;
    }

    #pragma unroll
    for (int t = 0; t < 2; ++t) {
        #pragma unroll
        for (int u = 0; u < 4; ++u) {
            v8f c = {};
            c = __builtin_amdgcn_wmma_f32_16x16x64_fp8_fp8(
                    afrag[t], bfrag[u], (short)0, c, false, false);
            float* orow = out + obase + (size_t)(m0 + t * 16 + 8 * hi) * NDIM
                              + (n0 + u * 16 + lr);
            #pragma unroll
            for (int r = 0; r < 8; ++r)
                orow[(size_t)r * NDIM] = c[r] * dq;
        }
    }
}

extern "C" void kernel_launch(void* const* d_in, const int* in_sizes, int n_in,
                              void* d_out, int out_size, void* d_ws, size_t ws_size,
                              hipStream_t stream) {
    const float* A = (const float*)d_in[0];
    const float* B = (const float*)d_in[1];
    float* out = (float*)d_out;

    float* amax = (float*)d_ws;
    unsigned char* Aq = (unsigned char*)d_ws + 256;
    unsigned char* Bq = Aq + (size_t)BATCHES * MDIM * KDIM;

    const int n4 = (BATCHES * MDIM * KDIM) / 4;

    init_amax<<<1, 32, 0, stream>>>(amax);
    amax_kernel<<<2048, 256, 0, stream>>>(A, amax + 0, n4);
    amax_kernel<<<2048, 256, 0, stream>>>(B, amax + 1, n4);
    quant_a<<<4096, 256, 0, stream>>>(A, (unsigned int*)Aq, amax + 0, n4);
    quant_b_transpose<<<dim3(NDIM / 64, BATCHES), 256, 0, stream>>>(B, Bq, amax + 1);
    gemm_fp8_wmma<<<dim3(MDIM / 128, NDIM / 128, BATCHES), 256, 0, stream>>>(Aq, Bq, out, amax);
}
